// RelationalConceptGraphLearning_80771154968584
// MI455X (gfx1250) — matrix operand order repaired and checked by reading, not compiled, not run on hardware
//
#include <hip/hip_runtime.h>
#include <cmath>

// Problem dims (compile-time)
#define Bb   16
#define Nn   1024
#define INs  100
#define INP  128   // IN padded to multiple of 32 for WMMA K
#define Hh   8
#define Dd   64
#define HD   512
#define OUTd 256

typedef __bf16 bf16_t;
typedef __attribute__((ext_vector_type(16))) __bf16 v16bf;
typedef __attribute__((ext_vector_type(8)))  float  v8f;

#define Z8 ((v8f){0.f,0.f,0.f,0.f,0.f,0.f,0.f,0.f})

static __device__ __forceinline__ v8f wmma_bf16(v16bf a, v16bf b, v8f c) {
  // D = A(16x32 bf16) * B(32x16 bf16) + C(16x16 f32)
  return __builtin_amdgcn_wmma_f32_16x16x32_bf16(false, a, false, b, (short)0, c,
                                                 false, false);
}

// ---- CDNA5 async memory->LDS (ASYNCcnt-tracked), CDNA5 ISA ------------
// Per-lane: copy 16 bytes global -> LDS. LDS address = low 32 bits of the
// generic pointer.
static __device__ __forceinline__ void async_load16(const bf16_t* gsrc, bf16_t* ldst) {
  unsigned lds_addr = (unsigned)(unsigned long long)(uintptr_t)ldst;
  unsigned long long ga = (unsigned long long)(uintptr_t)gsrc;
  asm volatile("global_load_async_to_lds_b128 %0, %1, off"
               :: "v"(lds_addr), "v"(ga) : "memory");
}
static __device__ __forceinline__ void wait_async0() {
  asm volatile("s_wait_asynccnt 0" ::: "memory");
}

// ---- Fragment loaders (m/n per CDNA5 ISA 7.12.2; k contiguous 16*half+s in A and B, wave32) ----------------

// A fragment: 16x32 bf16 tile from row-major src, leading dim ld (elements).
// lane l: row m=l&15, half=l>>4; VGPR j: k=half*16+2j(+1)
static __device__ __forceinline__ v16bf load_frag_a(const bf16_t* p, int ld) {
  const int lane = threadIdx.x & 31;
  const int m = lane & 15, hf = lane >> 4;
  const bf16_t* row = p + (size_t)m * ld;
  v16bf a;
#pragma unroll
  for (int j = 0; j < 8; ++j) {
    int k = hf * 16 + 2 * j;
    a[2 * j]     = row[k];
    a[2 * j + 1] = row[k + 1];
  }
  return a;
}

// B fragment where logical B[k][n] = src[n*ld + k]  (src stores B^T row-major;
// covers weight matrices W[n][k] and K-matrix rows for Q*K^T).
// lane l: col n=l&15, half=l>>4; VGPR v: k = half*16 + 2v (+1)  -> 32B contiguous
static __device__ __forceinline__ v16bf load_frag_bT(const bf16_t* p, int ld) {
  const int lane = threadIdx.x & 31;
  const int n = lane & 15, hf = lane >> 4;
  const bf16_t* col = p + (size_t)n * ld + hf * 16;
  v16bf b;
#pragma unroll
  for (int v = 0; v < 8; ++v) { b[2 * v] = col[2 * v]; b[2 * v + 1] = col[2 * v + 1]; }
  return b;
}

// B fragment from row-major B[k][n] (rows k, cols n contiguous), ld elements.
static __device__ __forceinline__ v16bf load_frag_b(const bf16_t* p, int ld) {
  const int lane = threadIdx.x & 31;
  const int n = lane & 15, hf = lane >> 4;
  v16bf b;
#pragma unroll
  for (int v = 0; v < 8; ++v) {
    b[2 * v]     = p[(size_t)(hf * 16 + 2 * v)     * ld + n];
    b[2 * v + 1] = p[(size_t)(hf * 16 + 2 * v + 1) * ld + n];
  }
  return b;
}

// ---- Conversion / padding ---------------------------------------------------

__global__ __launch_bounds__(256) void convert_pad_bf16(const float* __restrict__ src,
                                                        bf16_t* __restrict__ dst,
                                                        int rows, int cs, int cd) {
  int idx = blockIdx.x * blockDim.x + threadIdx.x;
  if (idx >= rows * cd) return;
  int r = idx / cd, c = idx % cd;
  dst[idx] = (c < cs) ? (bf16_t)src[(size_t)r * cs + c] : (bf16_t)0.0f;
}

// ---- Generic WMMA GEMM: C[M][Ncols] = act(A[M][K] * Bt[Ncols][K]^T + bias) --

__global__ __launch_bounds__(256) void gemm_bf16_wmma(const bf16_t* __restrict__ A,
                                                      const bf16_t* __restrict__ Bt,
                                                      const float* __restrict__ bias,
                                                      float* __restrict__ Cout,
                                                      int M, int Ncols, int K, int doRelu) {
  const int wave = threadIdx.x >> 5;
  const int tile = blockIdx.x * (blockDim.x >> 5) + wave;
  const int NT = Ncols / 16;
  if (tile >= (M / 16) * NT) return;              // wave-uniform branch
  const int mt = tile / NT, nt = tile % NT;

  v8f acc = Z8;
  const bf16_t* Arow = A + (size_t)mt * 16 * K;
  const bf16_t* Brow = Bt + (size_t)nt * 16 * K;
  for (int k = 0; k < K; k += 32) {
    __builtin_prefetch(Arow + k + 64, 0, 1);       // global_prefetch path
    v16bf af = load_frag_a(Arow + k, K);
    v16bf bf = load_frag_bT(Brow + k, K);
    acc = wmma_bf16(af, bf, acc);
  }
  const int lane = threadIdx.x & 31;
  const int nn = lane & 15, mb = (lane >> 4) * 8;
  const float bv = bias[nt * 16 + nn];
#pragma unroll
  for (int i = 0; i < 8; ++i) {
    float v = acc[i] + bv;
    if (doRelu) v = fmaxf(v, 0.0f);
    Cout[(size_t)(mt * 16 + mb + i) * Ncols + nt * 16 + nn] = v;
  }
}

// ---- Per-(b,h) LayerNorm stats over the full N*D slice ----------------------

__global__ __launch_bounds__(256) void bh_stats(const float* __restrict__ raw,
                                                float* __restrict__ stats) {
  const int bh = blockIdx.x;            // B*H blocks
  const int b = bh / Hh, h = bh % Hh;
  const float* base = raw + (size_t)b * Nn * HD + h * Dd;
  float s = 0.f, ss = 0.f;
  for (int i = threadIdx.x; i < Nn * Dd; i += blockDim.x) {
    int n = i / Dd, d = i % Dd;
    float v = base[(size_t)n * HD + d];
    s += v; ss += v * v;
  }
  __shared__ float sh0[256], sh1[256];
  sh0[threadIdx.x] = s; sh1[threadIdx.x] = ss; __syncthreads();
  for (int o = 128; o > 0; o >>= 1) {
    if ((int)threadIdx.x < o) { sh0[threadIdx.x] += sh0[threadIdx.x + o];
                                sh1[threadIdx.x] += sh1[threadIdx.x + o]; }
    __syncthreads();
  }
  if (threadIdx.x == 0) {
    float mean = sh0[0] / (float)(Nn * Dd);
    float var  = sh1[0] / (float)(Nn * Dd) - mean * mean;
    stats[bh * 2]     = mean;
    stats[bh * 2 + 1] = rsqrtf(var + 1e-5f);
  }
}

// normalize raw [B,N,HD] slice (b,h) -> bf16 [B,H,N,D] with affine g/b over (N,D)
__global__ __launch_bounds__(256) void qkv_norm(const float* __restrict__ raw,
                                                const float* __restrict__ g,
                                                const float* __restrict__ bb,
                                                const float* __restrict__ stats,
                                                bf16_t* __restrict__ out) {
  size_t idx = (size_t)blockIdx.x * blockDim.x + threadIdx.x;
  if (idx >= (size_t)Bb * Hh * Nn * Dd) return;
  int d = idx % Dd;
  int n = (idx / Dd) % Nn;
  int bh = idx / ((size_t)Nn * Dd);
  int b = bh / Hh, h = bh % Hh;
  float v = raw[((size_t)b * Nn + n) * HD + h * Dd + d];
  float y = (v - stats[bh * 2]) * stats[bh * 2 + 1] * g[n * Dd + d] + bb[n * Dd + d];
  out[idx] = (bf16_t)y;
}

// ---- Flash attention --------------------------------------------------------
// Block = 4 waves, one (b,h); waves own distinct 16-row Q tiles but SHARE the
// K/V stream: each 32-row K/V tile is staged once per block into LDS via
// async global->LDS DMA, then all waves read fragments from LDS.

__global__ __launch_bounds__(128) void flash_attn_wmma(const bf16_t* __restrict__ Qb,
                                                       const bf16_t* __restrict__ Kb,
                                                       const bf16_t* __restrict__ Vb,
                                                       bf16_t* __restrict__ E) {
  const int bh = blockIdx.y;
  const int b = bh / Hh, h = bh % Hh;
  const int tid = threadIdx.x;
  const int wave = tid >> 5;
  const int ft = blockIdx.x * 4 + wave;           // 0..63 row tiles
  const int lane = tid & 31;
  const int nn = lane & 15, hs = lane >> 4;

  const bf16_t* Q = Qb + (size_t)bh * Nn * Dd;
  const bf16_t* K = Kb + (size_t)bh * Nn * Dd;
  const bf16_t* V = Vb + (size_t)bh * Nn * Dd;

  __shared__ bf16_t Kt[32 * Dd];                  // 4KB staged K tile (row-major)
  __shared__ bf16_t Vt[32 * Dd];                  // 4KB staged V tile (row-major)
  __shared__ bf16_t Pt[4][16][32];                // per-wave P tile (C->A relayout)

  const v16bf aq0 = load_frag_a(Q + (size_t)ft * 16 * Dd, Dd);
  const v16bf aq1 = load_frag_a(Q + (size_t)ft * 16 * Dd + 32, Dd);

  v8f o0 = Z8, o1 = Z8, o2 = Z8, o3 = Z8;
  float mrow[8], lrow[8];
#pragma unroll
  for (int i = 0; i < 8; ++i) { mrow[i] = -3.0e38f; lrow[i] = 0.f; }

  for (int g2 = 0; g2 < Nn / 32; ++g2) {          // 32 kv rows per step
    __syncthreads();                              // prior readers done
    {
      const bf16_t* gK = K + (size_t)g2 * 32 * Dd;
      const bf16_t* gV = V + (size_t)g2 * 32 * Dd;
      // 4096B per tile = 256 x 16B chunks; 128 threads x 2 chunks each
      async_load16(gK + (size_t)tid * 8,         Kt + (size_t)tid * 8);
      async_load16(gK + (size_t)(tid + 128) * 8, Kt + (size_t)(tid + 128) * 8);
      async_load16(gV + (size_t)tid * 8,         Vt + (size_t)tid * 8);
      async_load16(gV + (size_t)(tid + 128) * 8, Vt + (size_t)(tid + 128) * 8);
      wait_async0();
    }
    __syncthreads();                              // staging visible to all waves

    // S = (Q * K^T) * 1/sqrt(D): two 16x16 score tiles (kv rows g2*32..+31)
    v8f s0 = Z8, s1 = Z8;
    s0 = wmma_bf16(aq0, load_frag_bT(Kt, Dd), s0);
    s0 = wmma_bf16(aq1, load_frag_bT(Kt + 32, Dd), s0);
    s1 = wmma_bf16(aq0, load_frag_bT(Kt + 16 * Dd, Dd), s1);
    s1 = wmma_bf16(aq1, load_frag_bT(Kt + 16 * Dd + 32, Dd), s1);

    float p0[8], p1[8], sc[8];
#pragma unroll
    for (int i = 0; i < 8; ++i) {
      float t0 = s0[i] * 0.125f, t1 = s1[i] * 0.125f;
      // row-wise reduce over the 16-lane group holding this row (masks<=8 stay in-group)
      float mx = fmaxf(t0, t1);
#pragma unroll
      for (int off = 8; off; off >>= 1) mx = fmaxf(mx, __shfl_xor(mx, off, 32));
      float mnew = fmaxf(mrow[i], mx);
      sc[i] = __expf(mrow[i] - mnew);
      mrow[i] = mnew;
      p0[i] = __expf(t0 - mnew);
      p1[i] = __expf(t1 - mnew);
      float rs = p0[i] + p1[i];
#pragma unroll
      for (int off = 8; off; off >>= 1) rs += __shfl_xor(rs, off, 32);
      lrow[i] = lrow[i] * sc[i] + rs;
    }
#pragma unroll
    for (int i = 0; i < 8; ++i) {
      o0[i] *= sc[i]; o1[i] *= sc[i]; o2[i] *= sc[i]; o3[i] *= sc[i];
    }
    // C-layout -> LDS row-major 16x32 (per-wave private; DS ops in-order per wave)
#pragma unroll
    for (int i = 0; i < 8; ++i) {
      int m = i + 8 * hs;
      Pt[wave][m][nn]      = (bf16_t)p0[i];
      Pt[wave][m][16 + nn] = (bf16_t)p1[i];
    }
    v16bf ap = load_frag_a(&Pt[wave][0][0], 32);
    // O += P(16x32) * V(32x64): 4 WMMAs over the D=64 column tiles
    o0 = wmma_bf16(ap, load_frag_b(Vt +  0, Dd), o0);
    o1 = wmma_bf16(ap, load_frag_b(Vt + 16, Dd), o1);
    o2 = wmma_bf16(ap, load_frag_b(Vt + 32, Dd), o2);
    o3 = wmma_bf16(ap, load_frag_b(Vt + 48, Dd), o3);
  }

  // normalize rows by softmax denominator and scatter to E[b][n][h*64 + d] (bf16)
#pragma unroll
  for (int i = 0; i < 8; ++i) {
    int m = i + 8 * hs;
    float inv = 1.0f / lrow[i];
    size_t row = ((size_t)b * Nn + ft * 16 + m) * HD + h * Dd;
    E[row +  0 + nn] = (bf16_t)(o0[i] * inv);
    E[row + 16 + nn] = (bf16_t)(o1[i] * inv);
    E[row + 32 + nn] = (bf16_t)(o2[i] * inv);
    E[row + 48 + nn] = (bf16_t)(o3[i] * inv);
  }
}

// ---- Post-FF1 reductions: LN(no affine) folded into max-pool ----------------

__global__ __launch_bounds__(256) void f_stats(const float* __restrict__ F,
                                               float* __restrict__ stats) {
  const int b = blockIdx.x;
  const float* base = F + (size_t)b * Nn * HD;
  float s = 0.f, ss = 0.f;
  for (int i = threadIdx.x; i < Nn * HD; i += blockDim.x) {
    float v = base[i]; s += v; ss += v * v;
  }
  __shared__ float sh0[256], sh1[256];
  sh0[threadIdx.x] = s; sh1[threadIdx.x] = ss; __syncthreads();
  for (int o = 128; o > 0; o >>= 1) {
    if ((int)threadIdx.x < o) { sh0[threadIdx.x] += sh0[threadIdx.x + o];
                                sh1[threadIdx.x] += sh1[threadIdx.x + o]; }
    __syncthreads();
  }
  if (threadIdx.x == 0) {
    float mean = sh0[0] / (float)(Nn * HD);
    float var  = sh1[0] / (float)(Nn * HD) - mean * mean;
    stats[b * 2]     = mean;
    stats[b * 2 + 1] = rsqrtf(var + 1e-5f);
  }
}

__global__ __launch_bounds__(256) void f_colmax(const float* __restrict__ F,
                                                float* __restrict__ colmax) {
  int idx = blockIdx.x * blockDim.x + threadIdx.x;
  if (idx >= Bb * HD) return;
  int b = idx / HD, j = idx % HD;
  float mx = -3.0e38f;
  for (int n = 0; n < Nn; ++n) mx = fmaxf(mx, F[((size_t)b * Nn + n) * HD + j]);
  colmax[idx] = mx;
}

// y = elu( ((colmax - mean)*rstd) @ W2^T + b2 )
__global__ __launch_bounds__(256) void final_head(const float* __restrict__ colmax,
                                                  const float* __restrict__ fstats,
                                                  const float* __restrict__ W2,
                                                  const float* __restrict__ b2,
                                                  float* __restrict__ y) {
  int idx = blockIdx.x * blockDim.x + threadIdx.x;
  if (idx >= Bb * OUTd) return;
  int b = idx / OUTd, o = idx % OUTd;
  float mean = fstats[b * 2], rstd = fstats[b * 2 + 1];
  float acc = b2[o];
  const float* cm = colmax + (size_t)b * HD;
  const float* wr = W2 + (size_t)o * HD;
  for (int j = 0; j < HD; ++j) acc += (cm[j] - mean) * rstd * wr[j];
  y[idx] = (acc > 0.f) ? acc : expm1f(acc);
}

// ---- Host orchestration -----------------------------------------------------

extern "C" void kernel_launch(void* const* d_in, const int* in_sizes, int n_in,
                              void* d_out, int out_size, void* d_ws, size_t ws_size,
                              hipStream_t stream) {
  (void)in_sizes; (void)n_in; (void)out_size; (void)ws_size;
  const float* x  = (const float*)d_in[0];
  const float* Wk = (const float*)d_in[1];
  const float* bk = (const float*)d_in[2];
  const float* Wq = (const float*)d_in[3];
  const float* bq = (const float*)d_in[4];
  const float* Wv = (const float*)d_in[5];
  const float* bv = (const float*)d_in[6];
  const float* kg = (const float*)d_in[7];
  const float* kbb= (const float*)d_in[8];
  const float* qg = (const float*)d_in[9];
  const float* qbb= (const float*)d_in[10];
  const float* vg = (const float*)d_in[11];
  const float* vbb= (const float*)d_in[12];
  const float* W1 = (const float*)d_in[13];
  const float* b1 = (const float*)d_in[14];
  const float* W2 = (const float*)d_in[15];
  const float* b2 = (const float*)d_in[16];
  float* y = (float*)d_out;

  char* ws = (char*)d_ws;
  size_t off = 0;
  auto take = [&](size_t bytes) { size_t o = off; off += (bytes + 255) & ~(size_t)255; return o; };

  const size_t M = (size_t)Bb * Nn;                 // 16384 rows
  size_t xb_off  = take(M * INP * 2);               // x in bf16, padded
  size_t wkb_off = take((size_t)HD * INP * 2);
  size_t wqb_off = take((size_t)HD * INP * 2);
  size_t wvb_off = take((size_t)HD * INP * 2);
  size_t w1b_off = take((size_t)HD * HD * 2);
  size_t raw_off = take(M * HD * 4);                // fp32 proj scratch; later E (bf16)
  size_t qkv_off = take((size_t)3 * Bb * Hh * Nn * Dd * 2); // later F (fp32) overlaps
  size_t sbh_off = take((size_t)Bb * Hh * 2 * 4);
  size_t fst_off = take((size_t)Bb * 2 * 4);
  size_t cmx_off = take((size_t)Bb * HD * 4);

  bf16_t* xb  = (bf16_t*)(ws + xb_off);
  bf16_t* wkb = (bf16_t*)(ws + wkb_off);
  bf16_t* wqb = (bf16_t*)(ws + wqb_off);
  bf16_t* wvb = (bf16_t*)(ws + wvb_off);
  bf16_t* w1b = (bf16_t*)(ws + w1b_off);
  float*  raw = (float*)(ws + raw_off);
  bf16_t* qkvb= (bf16_t*)(ws + qkv_off);
  float*  sbh = (float*)(ws + sbh_off);
  float*  fst = (float*)(ws + fst_off);
  float*  cmx = (float*)(ws + cmx_off);

  const size_t perMat = (size_t)Bb * Hh * Nn * Dd;  // bf16 elements per Q/K/V
  bf16_t* Kn = qkvb;                 // normalized K [B,H,N,D]
  bf16_t* Qn = qkvb + perMat;
  bf16_t* Vn = qkvb + 2 * perMat;
  bf16_t* Eb = (bf16_t*)(ws + raw_off);             // reuse raw region (16.8MB<=33.5MB)
  float*  F  = (float*)(ws + qkv_off);              // reuse qkv region (33.5MB<=50.3MB)

  // 1) convert/pad activations + weights to bf16
  convert_pad_bf16<<<(int)((M * INP + 255) / 256), 256, 0, stream>>>(x,  xb,  (int)M, INs, INP);
  convert_pad_bf16<<<(HD * INP + 255) / 256, 256, 0, stream>>>(Wk, wkb, HD, INs, INP);
  convert_pad_bf16<<<(HD * INP + 255) / 256, 256, 0, stream>>>(Wq, wqb, HD, INs, INP);
  convert_pad_bf16<<<(HD * INP + 255) / 256, 256, 0, stream>>>(Wv, wvb, HD, INs, INP);
  convert_pad_bf16<<<(HD * HD + 255) / 256, 256, 0, stream>>>(W1, w1b, HD, HD, HD);

  // 2) projections (WMMA) + per-(b,h) LN + relayout to [B,H,N,D] bf16
  const int projTiles = (int)(M / 16) * (HD / 16);  // 32768 tiles, 8 waves/block
  const int projBlocks = projTiles / 8;
  const int normTotal = (int)perMat;
  struct { const bf16_t* w; const float* bias; const float* g; const float* b; bf16_t* out; } mats[3] =
    { {wkb, bk, kg, kbb, Kn}, {wqb, bq, qg, qbb, Qn}, {wvb, bv, vg, vbb, Vn} };
  for (int p = 0; p < 3; ++p) {
    gemm_bf16_wmma<<<projBlocks, 256, 0, stream>>>(xb, mats[p].w, mats[p].bias, raw,
                                                   (int)M, HD, INP, 0);
    bh_stats<<<Bb * Hh, 256, 0, stream>>>(raw, sbh);
    qkv_norm<<<(normTotal + 255) / 256, 256, 0, stream>>>(raw, mats[p].g, mats[p].b, sbh,
                                                          mats[p].out);
  }

  // 3) flash attention (WMMA + async LDS staging), E bf16 [B,N,HD]
  flash_attn_wmma<<<dim3(16, Bb * Hh), 128, 0, stream>>>(Qn, Kn, Vn, Eb);

  // 4) FF1 (WMMA) + ReLU -> F fp32 [B,N,HD]  (qkv region dead -> reused)
  gemm_bf16_wmma<<<projBlocks, 256, 0, stream>>>(Eb, w1b, b1, F, (int)M, HD, HD, 1);

  // 5) LN-over-[N,HD] stats + column max (LN folds through max-pool)
  f_stats<<<Bb, 256, 0, stream>>>(F, fst);
  f_colmax<<<(Bb * HD + 255) / 256, 256, 0, stream>>>(F, cmx);

  // 6) final linear + ELU
  final_head<<<(Bb * OUTd + 255) / 256, 256, 0, stream>>>(cmx, fst, W2, b2, y);
}
